// SelfAttention_15573551415482
// MI455X (gfx1250) — matrix operand order repaired and checked by reading, not compiled, not run on hardware
//
#include <hip/hip_runtime.h>

// ---------------------------------------------------------------------------
// Self-attention on MI455X (gfx1250): bf16 WMMA GEMMs with f32 accumulation,
// Tensor Data Mover (TDM) staging for the pure-bf16 tiles.
// B=4, S=2048, D=1024.
// ---------------------------------------------------------------------------

#define S_LEN   2048
#define D_DIM   1024
#define BATCH   4
#define M_ROWS  (BATCH * S_LEN)        // 8192 rows for the QKV projection
#define LDA     40                     // LDS row pitch (bf16 elems): 80B, 16B-aligned

typedef __attribute__((ext_vector_type(16))) __bf16 bf16x16;
typedef __attribute__((ext_vector_type(8)))  __bf16 bf16x8;
typedef __attribute__((ext_vector_type(8)))  float  f32x8;
typedef __attribute__((ext_vector_type(4)))  unsigned int u32x4;
typedef __attribute__((ext_vector_type(8)))  int  i32x8;
typedef __attribute__((ext_vector_type(4)))  int  i32x4;

// float -> bf16 (round to nearest even), bit-level so we control rounding.
__device__ __forceinline__ unsigned short f2bf(float f) {
    unsigned int u = __float_as_uint(f);
    u += 0x7FFFu + ((u >> 16) & 1u);
    return (unsigned short)(u >> 16);
}

// Build a 16-element bf16 fragment from two 16-byte LDS chunks.
__device__ __forceinline__ bf16x16 frag16(const unsigned short* sh, int off0, int off1) {
    bf16x8 lo = *reinterpret_cast<const bf16x8*>(sh + off0);
    bf16x8 hi = *reinterpret_cast<const bf16x8*>(sh + off1);
    return __builtin_shufflevector(lo, hi, 0,1,2,3,4,5,6,7,8,9,10,11,12,13,14,15);
}

__device__ __forceinline__ f32x8 wmma_bf16(bf16x16 a, bf16x16 b, f32x8 c) {
    return __builtin_amdgcn_wmma_f32_16x16x32_bf16(false, a, false, b, (short)0, c,
                                                   false, false);
}

// ---------------------------------------------------------------------------
// TDM: DMA one 128-row x 32-col bf16 tile from global memory into LDS,
// padding each 64B row with 16B so the LDS pitch is LDA=40 bf16 elements.
//   lds_off      : LDS byte offset of destination tile
//   gptr         : global address of tile start (bf16 elements)
//   tensor_w/h   : tensor extent (elements / rows), for OOB clamping
//   stride_elems : row pitch of the source tensor in bf16 elements
// D# layout per CDNA5 ISA ch.8 (group0 128b, group1 256b; groups 2/3 unused
// for 2-D tensors and passed as zeros).  This toolchain's builtin takes the
// 6-arg form: (g0, g1, g2, g3, g4, cpol).
// ---------------------------------------------------------------------------
__device__ __forceinline__ void tdm_load_tile_bf16(unsigned lds_off,
                                                   const unsigned short* gptr,
                                                   unsigned tensor_w,
                                                   unsigned tensor_h,
                                                   unsigned stride_elems) {
    unsigned long long ga = (unsigned long long)(uintptr_t)gptr;
    u32x4 g0;
    g0[0] = 1u;                                               // count=1 (valid), user D#
    g0[1] = lds_off;                                          // lds_addr (bytes)
    g0[2] = (unsigned)ga;                                     // global_addr[31:0]
    g0[3] = ((unsigned)(ga >> 32) & 0x01FFFFFFu) | (2u << 30);// addr[56:32] | type=2

    i32x8 g1;
    // data_size=1 (2B) | pad_enable | pad_interval=3 (16 DW) | pad_amount=3 (4 DW)
    g1[0] = (int)((1u << 16) | (1u << 20) | (3u << 22) | (3u << 25));
    g1[1] = (int)((tensor_w & 0xFFFFu) << 16);                // tensor_dim0[15:0]
    g1[2] = (int)(((tensor_w >> 16) & 0xFFFFu) |              // tensor_dim0[31:16]
                  ((tensor_h & 0xFFFFu) << 16));              // tensor_dim1[15:0]
    g1[3] = (int)(((tensor_h >> 16) & 0xFFFFu) |              // tensor_dim1[31:16]
                  (32u << 16));                               // tile_dim0 = 32
    g1[4] = (int)128u;                                        // tile_dim1 = 128
    g1[5] = (int)stride_elems;                                // tensor_dim0_stride[31:0]
    g1[6] = 0;                                                // stride hi / dim1_stride lo
    g1[7] = 0;

    i32x4 z4 = {0, 0, 0, 0};
    i32x8 z8 = {0, 0, 0, 0, 0, 0, 0, 0};
    __builtin_amdgcn_tensor_load_to_lds(g0, g1, z4, z4, z8, 0);
}

// Uniform "am I wave 0 of this workgroup" predicate that forces a scalar
// branch (TDM ignores EXEC, so exec-masking alone would not gate it).
__device__ __forceinline__ bool wave0_scalar() {
    return (__builtin_amdgcn_readfirstlane(threadIdx.x) >> 5) == 0;
}

__device__ __forceinline__ unsigned lds_offset_of(const void* p) {
    return (unsigned)(uintptr_t)p;   // flat LDS mapping: offset lives in addr[31:0]
}

// ---------------------------------------------------------------------------
// Kernel 1: QKV projection.  out[z] = x @ W[z]   (f32 inputs -> bf16 outputs)
// Grid: (M/128, N/128, 3), block 256.  Tile 128x128, K-step 32.
// f32->bf16 conversion happens on the staging path, so tiles stay on VALU.
// ---------------------------------------------------------------------------
__global__ void __launch_bounds__(256)
qkv_gemm(const float* __restrict__ x,
         const float* __restrict__ wq,
         const float* __restrict__ wk,
         const float* __restrict__ wv,
         unsigned short* __restrict__ qkv) {
    __shared__ __align__(16) unsigned short Ash[128 * LDA];
    __shared__ __align__(16) unsigned short Bsh[128 * LDA];

    const int m0 = blockIdx.x * 128;
    const int n0 = blockIdx.y * 128;
    const int z  = blockIdx.z;
    const float* w = (z == 0) ? wq : (z == 1) ? wk : wv;
    unsigned short* outp = qkv + (size_t)z * M_ROWS * D_DIM;

    const int tid   = threadIdx.x;
    const int lane  = tid & 31;
    const int wid   = tid >> 5;
    const int waveM = wid & 3;          // 4 waves along M
    const int waveN = wid >> 2;         // 2 waves along N
    const int am    = lane & 15;
    const int ah    = lane >> 4;

    f32x8 acc[2][4] = {};

    for (int kb = 0; kb < D_DIM; kb += 32) {
        // A tile: x[m0..+128, kb..+32]  -> bf16, row-major
        for (int idx = tid; idx < 128 * 32; idx += 256) {
            int r = idx >> 5, c = idx & 31;
            Ash[r * LDA + c] = f2bf(x[(size_t)(m0 + r) * D_DIM + kb + c]);
        }
        // B tile (transposed): w[kb..+32, n0..+128] -> Bsh[n][k]
        for (int idx = tid; idx < 32 * 128; idx += 256) {
            int r = idx >> 7, c = idx & 127;   // r = k, c = n
            Bsh[c * LDA + r] = f2bf(w[(size_t)(kb + r) * D_DIM + n0 + c]);
        }
        __syncthreads();

        bf16x16 afrag[2], bfrag[4];
#pragma unroll
        for (int mt = 0; mt < 2; ++mt) {
            int row = (waveM * 32 + mt * 16 + am) * LDA + ah * 8;
            afrag[mt] = frag16(Ash, row, row + 16);
        }
#pragma unroll
        for (int nt = 0; nt < 4; ++nt) {
            int row = (waveN * 64 + nt * 16 + am) * LDA + ah * 8;
            bfrag[nt] = frag16(Bsh, row, row + 16);
        }
#pragma unroll
        for (int mt = 0; mt < 2; ++mt)
#pragma unroll
            for (int nt = 0; nt < 4; ++nt)
                acc[mt][nt] = wmma_bf16(afrag[mt], bfrag[nt], acc[mt][nt]);
        __syncthreads();
    }

#pragma unroll
    for (int mt = 0; mt < 2; ++mt)
#pragma unroll
        for (int nt = 0; nt < 4; ++nt) {
            int baseRow = m0 + waveM * 32 + mt * 16 + ah * 8;
            int col     = n0 + waveN * 64 + nt * 16 + am;
#pragma unroll
            for (int r = 0; r < 8; ++r)
                outp[(size_t)(baseRow + r) * D_DIM + col] = f2bf(acc[mt][nt][r]);
        }
}

// ---------------------------------------------------------------------------
// Kernel 2: scores = (Q @ K^T) / 32, masked with -inf.  Per batch.
// Both operands are bf16 row-major [S, D]; both tiles are straight copies,
// so both are staged by the Tensor Data Mover (wave 0 issues, TENSORcnt).
// Grid: (16, 16), block 256.
// ---------------------------------------------------------------------------
__global__ void __launch_bounds__(256)
scores_gemm(const unsigned short* __restrict__ Q,
            const unsigned short* __restrict__ K,
            const unsigned char* __restrict__ mask,
            float* __restrict__ scores) {
    __shared__ __align__(16) unsigned short Ash[128 * LDA];
    __shared__ __align__(16) unsigned short Bsh[128 * LDA];

    const int m0 = blockIdx.x * 128;    // query rows
    const int n0 = blockIdx.y * 128;    // key rows
    const int tid   = threadIdx.x;
    const int lane  = tid & 31;
    const int wid   = tid >> 5;
    const int waveM = wid & 3;
    const int waveN = wid >> 2;
    const int am    = lane & 15;
    const int ah    = lane >> 4;

    const bool issuer = wave0_scalar();
    const unsigned ashOff = lds_offset_of(Ash);
    const unsigned bshOff = lds_offset_of(Bsh);

    f32x8 acc[2][4] = {};

    for (int kb = 0; kb < D_DIM; kb += 32) {
        if (issuer) {
            tdm_load_tile_bf16(ashOff, Q + (size_t)m0 * D_DIM + kb, D_DIM, S_LEN, D_DIM);
            tdm_load_tile_bf16(bshOff, K + (size_t)n0 * D_DIM + kb, D_DIM, S_LEN, D_DIM);
            __builtin_amdgcn_s_wait_tensorcnt(0);
        }
        __syncthreads();

        bf16x16 afrag[2], bfrag[4];
#pragma unroll
        for (int mt = 0; mt < 2; ++mt) {
            int row = (waveM * 32 + mt * 16 + am) * LDA + ah * 8;
            afrag[mt] = frag16(Ash, row, row + 16);
        }
#pragma unroll
        for (int nt = 0; nt < 4; ++nt) {
            int row = (waveN * 64 + nt * 16 + am) * LDA + ah * 8;
            bfrag[nt] = frag16(Bsh, row, row + 16);
        }
#pragma unroll
        for (int mt = 0; mt < 2; ++mt)
#pragma unroll
            for (int nt = 0; nt < 4; ++nt)
                acc[mt][nt] = wmma_bf16(afrag[mt], bfrag[nt], acc[mt][nt]);
        __syncthreads();
    }

    const float kNegInf = -__builtin_inff();
#pragma unroll
    for (int mt = 0; mt < 2; ++mt)
#pragma unroll
        for (int nt = 0; nt < 4; ++nt) {
            int baseRow = m0 + waveM * 32 + mt * 16 + ah * 8;
            int col     = n0 + waveN * 64 + nt * 16 + am;
#pragma unroll
            for (int r = 0; r < 8; ++r) {
                size_t off = (size_t)(baseRow + r) * S_LEN + col;
                float v = acc[mt][nt][r] * 0.03125f;     // 1/sqrt(1024)
                if (mask[off]) v = kNegInf;
                scores[off] = v;
            }
        }
}

// ---------------------------------------------------------------------------
// Kernel 3: row softmax over 2048 f32 scores; writes bf16 probabilities
// IN PLACE (first 4 KB of each 8 KB row).  Grid: 2048 blocks of 256.
// ---------------------------------------------------------------------------
__global__ void __launch_bounds__(256)
softmax_rows(float* __restrict__ scores) {
    __shared__ float red[256];
    const int tid = threadIdx.x;
    float* rp = scores + (size_t)blockIdx.x * S_LEN;

    float v[8];
    float mx = -__builtin_inff();
#pragma unroll
    for (int i = 0; i < 8; ++i) {
        v[i] = rp[tid + (i << 8)];
        mx = fmaxf(mx, v[i]);
    }
    red[tid] = mx;
    __syncthreads();
    for (int s = 128; s > 0; s >>= 1) {
        if (tid < s) red[tid] = fmaxf(red[tid], red[tid + s]);
        __syncthreads();
    }
    mx = red[0];
    __syncthreads();

    float e[8], sum = 0.f;
#pragma unroll
    for (int i = 0; i < 8; ++i) {
        e[i] = __expf(v[i] - mx);
        sum += e[i];
    }
    red[tid] = sum;
    __syncthreads();
    for (int s = 128; s > 0; s >>= 1) {
        if (tid < s) red[tid] += red[tid + s];
        __syncthreads();
    }
    const float inv = 1.0f / red[0];

    unsigned short* wp = reinterpret_cast<unsigned short*>(rp);
#pragma unroll
    for (int i = 0; i < 8; ++i)
        wp[tid + (i << 8)] = f2bf(e[i] * inv);
}

// ---------------------------------------------------------------------------
// Kernel 4: out = P @ V.  P: bf16, row pitch 4096 elems (in-place over f32
// score rows) -> TDM-staged.  V needs a transpose, so it stays on the VALU
// path.  Grid: (16, 8), block 256.
// ---------------------------------------------------------------------------
__global__ void __launch_bounds__(256)
av_gemm(const unsigned short* __restrict__ P,
        const unsigned short* __restrict__ V,
        float* __restrict__ out) {
    __shared__ __align__(16) unsigned short Ash[128 * LDA];
    __shared__ __align__(16) unsigned short Bsh[128 * LDA];

    const int m0 = blockIdx.x * 128;    // query rows
    const int n0 = blockIdx.y * 128;    // head dims
    const int tid   = threadIdx.x;
    const int lane  = tid & 31;
    const int wid   = tid >> 5;
    const int waveM = wid & 3;
    const int waveN = wid >> 2;
    const int am    = lane & 15;
    const int ah    = lane >> 4;

    const bool issuer = wave0_scalar();
    const unsigned ashOff = lds_offset_of(Ash);

    f32x8 acc[2][4] = {};

    for (int kb = 0; kb < S_LEN; kb += 32) {
        // A tile via TDM: probability rows, pitch 4096 bf16 elements
        if (issuer) {
            tdm_load_tile_bf16(ashOff, P + (size_t)m0 * 4096 + kb, 4096, S_LEN, 4096);
        }
        // B tile (transposed): V[kb..+32, n0..+128] -> Bsh[n][k]
        for (int idx = tid; idx < 32 * 128; idx += 256) {
            int r = idx >> 7, c = idx & 127;   // r = k, c = n
            Bsh[c * LDA + r] = V[(size_t)(kb + r) * D_DIM + n0 + c];
        }
        if (issuer) {
            __builtin_amdgcn_s_wait_tensorcnt(0);
        }
        __syncthreads();

        bf16x16 afrag[2], bfrag[4];
#pragma unroll
        for (int mt = 0; mt < 2; ++mt) {
            int row = (waveM * 32 + mt * 16 + am) * LDA + ah * 8;
            afrag[mt] = frag16(Ash, row, row + 16);
        }
#pragma unroll
        for (int nt = 0; nt < 4; ++nt) {
            int row = (waveN * 64 + nt * 16 + am) * LDA + ah * 8;
            bfrag[nt] = frag16(Bsh, row, row + 16);
        }
#pragma unroll
        for (int mt = 0; mt < 2; ++mt)
#pragma unroll
            for (int nt = 0; nt < 4; ++nt)
                acc[mt][nt] = wmma_bf16(afrag[mt], bfrag[nt], acc[mt][nt]);
        __syncthreads();
    }

#pragma unroll
    for (int mt = 0; mt < 2; ++mt)
#pragma unroll
        for (int nt = 0; nt < 4; ++nt) {
            int baseRow = m0 + waveM * 32 + mt * 16 + ah * 8;
            int col     = n0 + waveN * 64 + nt * 16 + am;
#pragma unroll
            for (int r = 0; r < 8; ++r)
                out[(size_t)(baseRow + r) * D_DIM + col] = acc[mt][nt][r];
        }
}

// ---------------------------------------------------------------------------
// Host launcher.
// Workspace layout (64 MB):
//   [0, 48MB)  : bf16 Q | K | V, each [8192, 1024]
//   [48, 64MB) : one batch of f32 scores [2048, 2048], reused per batch and
//                rewritten in place as bf16 probabilities by softmax.
// ---------------------------------------------------------------------------
extern "C" void kernel_launch(void* const* d_in, const int* in_sizes, int n_in,
                              void* d_out, int out_size, void* d_ws, size_t ws_size,
                              hipStream_t stream) {
    (void)in_sizes; (void)n_in; (void)out_size; (void)ws_size;

    const float*         x    = (const float*)d_in[0];
    const unsigned char* mask = (const unsigned char*)d_in[1];
    const float*         wq   = (const float*)d_in[2];
    const float*         wk   = (const float*)d_in[3];
    const float*         wv   = (const float*)d_in[4];
    float*               out  = (float*)d_out;

    const size_t QSZ = (size_t)M_ROWS * D_DIM;          // elems per projection
    unsigned short* qkv    = (unsigned short*)d_ws;
    float*          scores = (float*)((char*)d_ws + 3 * QSZ * sizeof(unsigned short));

    // 1) QKV projections: 3 GEMMs fused into one launch over grid.z
    qkv_gemm<<<dim3(M_ROWS / 128, D_DIM / 128, 3), 256, 0, stream>>>(x, wq, wk, wv, qkv);

    // 2-4) per-batch attention; batches serialize on the stream so the 16 MB
    // score tile is reused (it lives in L2 the whole time).
    for (int b = 0; b < BATCH; ++b) {
        const unsigned short* Qb = qkv + (size_t)b * S_LEN * D_DIM;
        const unsigned short* Kb = qkv + QSZ + (size_t)b * S_LEN * D_DIM;
        const unsigned short* Vb = qkv + 2 * QSZ + (size_t)b * S_LEN * D_DIM;

        scores_gemm<<<dim3(S_LEN / 128, S_LEN / 128), 256, 0, stream>>>(Qb, Kb, mask, scores);
        softmax_rows<<<S_LEN, 256, 0, stream>>>(scores);
        av_gemm<<<dim3(S_LEN / 128, D_DIM / 128), 256, 0, stream>>>(
            (const unsigned short*)scores, Vb, out + (size_t)b * S_LEN * D_DIM);
    }
}
